// LSTMAutoencoder_23106924052888
// MI455X (gfx1250) — compile-verified, matrix operand order checked by reading, not run on hardware
//
#include <hip/hip_runtime.h>
#include <hip/hip_bf16.h>

// ---------------------------------------------------------------------------
// LSTM autoencoder, B=128 T=1024 D=64 H=256 L=2, bf16 WMMA (f32 accumulate)
// ---------------------------------------------------------------------------

typedef __attribute__((ext_vector_type(8)))  __bf16 v8bf;
typedef __attribute__((ext_vector_type(16))) __bf16 v16bf;
typedef __attribute__((ext_vector_type(8)))  float  v8f;

#define B_DIM 128
#define T_DIM 1024
#define D_DIM 64
#define H_DIM 256
#define G4H   1024   // 4*H

// ---- WMMA helpers ---------------------------------------------------------

// 16x32 bf16 fragment (A or B operand), per ISA 7.12.2 16-bit layout:
//   lanes 0-15 : row = lane,    K chunks {k0..k0+7, k0+16..k0+23}
//   lanes 16-31: row = lane-16, K chunks {k0+8..k0+15, k0+24..k0+31}
__device__ __forceinline__ v16bf load_frag_16x32(const __bf16* __restrict__ base,
                                                 int row0, int ld, int k0, int lane) {
  const int half = lane >> 4;
  const int r    = lane & 15;
  const __bf16* p = base + (size_t)(row0 + r) * ld + (k0 + half * 8);
  v8bf lo = *(const v8bf*)(p);
  v8bf hi = *(const v8bf*)(p + 16);
  return __builtin_shufflevector(lo, hi, 0, 1, 2, 3, 4, 5, 6, 7,
                                         8, 9, 10, 11, 12, 13, 14, 15);
}

__device__ __forceinline__ v8f wmma_bf16(v16bf a, v16bf b, v8f c) {
  return __builtin_amdgcn_wmma_f32_16x16x32_bf16(false, a, false, b,
                                                 (short)0, c, false, false);
}

__device__ __forceinline__ float fast_sig(float x) {
  return 1.0f / (1.0f + __expf(-x));
}
__device__ __forceinline__ float fast_tanh(float x) {
  float e = __expf(2.0f * x);
  return 1.0f - 2.0f / (e + 1.0f);
}

// ---- small elementwise kernels -------------------------------------------

__global__ void cvt_f32_bf16(const float* __restrict__ in, __bf16* __restrict__ out, int n) {
  int i = blockIdx.x * blockDim.x + threadIdx.x;
  if (i < n) out[i] = (__bf16)in[i];
}

__global__ void combine_bias(const float* __restrict__ a, const float* __restrict__ b,
                             float* __restrict__ out, int n) {
  int i = blockIdx.x * blockDim.x + threadIdx.x;
  if (i < n) out[i] = a[i] + b[i];
}

__global__ void zero_bf16(__bf16* p, int n) {
  int i = blockIdx.x * blockDim.x + threadIdx.x;
  if (i < n) p[i] = (__bf16)0.0f;
}
__global__ void zero_f32(float* p, int n) {
  int i = blockIdx.x * blockDim.x + threadIdx.x;
  if (i < n) p[i] = 0.0f;
}
__global__ void init_barrier(unsigned* p) {
  if (threadIdx.x == 0) { p[0] = 0u; p[1] = 0u; }
}

// ---- projection GEMM: out[m,n] = sum_k A[m,k]*W[n,k] + bias[n] ------------
// A: [M,K] bf16 row-major, W: [N,K] bf16 row-major.
// Block = 128 threads (4 waves stacked in M); each wave owns a 32x32 output
// tile (2x2 register blocking: 2 A-frags x 2 B-frags -> 4 WMMAs / 4 loads
// per k-step). Grid: (M/128, N/32).

template <bool OUT_F32>
__global__ __launch_bounds__(128)
void proj_gemm(const __bf16* __restrict__ A, const __bf16* __restrict__ W,
               const float* __restrict__ bias,
               __bf16* __restrict__ outb, float* __restrict__ outf,
               int K, int N) {
  const int lane = threadIdx.x & 31;
  const int wave = threadIdx.x >> 5;
  const int m0 = blockIdx.x * 128 + wave * 32;
  const int n0 = blockIdx.y * 32;

  v8f acc00 = {}, acc01 = {}, acc10 = {}, acc11 = {};
  for (int k0 = 0; k0 < K; k0 += 32) {
    v16bf a0 = load_frag_16x32(A, m0,      K, k0, lane);
    v16bf a1 = load_frag_16x32(A, m0 + 16, K, k0, lane);
    v16bf b0 = load_frag_16x32(W, n0,      K, k0, lane);
    v16bf b1 = load_frag_16x32(W, n0 + 16, K, k0, lane);
    acc00 = wmma_bf16(a0, b0, acc00);
    acc01 = wmma_bf16(a0, b1, acc01);
    acc10 = wmma_bf16(a1, b0, acc10);
    acc11 = wmma_bf16(a1, b1, acc11);
  }

  const int half = lane >> 4;
  const int c    = lane & 15;
  const float bn0 = bias[n0 + c];
  const float bn1 = bias[n0 + 16 + c];
#pragma unroll
  for (int r = 0; r < 8; ++r) {
    const size_t mA = (size_t)(m0 + half * 8 + r);
    const size_t mB = mA + 16;
    if (OUT_F32) {
      outf[mA * N + n0 + c]      = acc00[r] + bn0;
      outf[mA * N + n0 + 16 + c] = acc01[r] + bn1;
      outf[mB * N + n0 + c]      = acc10[r] + bn0;
      outf[mB * N + n0 + 16 + c] = acc11[r] + bn1;
    } else {
      outb[mA * N + n0 + c]      = (__bf16)(acc00[r] + bn0);
      outb[mA * N + n0 + 16 + c] = (__bf16)(acc01[r] + bn1);
      outb[mB * N + n0 + c]      = (__bf16)(acc10[r] + bn0);
      outb[mB * N + n0 + 16 + c] = (__bf16)(acc11[r] + bn1);
    }
  }
}

// ---- device-scope grid barrier (sense via generation counter) -------------

__device__ __forceinline__ void grid_barrier(unsigned* cnt, unsigned* gen, unsigned nwg) {
  __syncthreads();
  if (threadIdx.x == 0) {
    __threadfence();
    volatile unsigned* vgen = (volatile unsigned*)gen;
    unsigned g = *vgen;
    if (atomicAdd(cnt, 1u) == nwg - 1u) {
      *cnt = 0u;
      __threadfence();
      atomicAdd(gen, 1u);
    } else {
      while (*vgen == g) { __builtin_amdgcn_s_sleep(1); }
    }
  }
  __syncthreads();
  __threadfence();
}

// ---- persistent LSTM scan over T steps ------------------------------------
// Grid: (B/16, H/32) = (8,8) = 64 WGs (fewer barrier participants), block =
// 256 threads = 8 waves; wave w: gate = w>>1, j-tile = (w&1)*16 within the
// WG's 32-wide hidden slice. Each wave's Whh B-fragments (K=256 -> 8 frags,
// 64 VGPRs) are preloaded and live in registers for the entire scan. h state
// is double-buffered bf16 (T even -> final state lands back in buffer 0).
// xg addressing: xg[b*bStride + t*tStride + n]; broadcast decoder input uses
// (bStride=4H, tStride=0). Next timestep's xg rows are prefetched
// (global_prefetch_b8) while the current step's WMMAs run.

__global__ __launch_bounds__(256)
void lstm_scan(const __bf16* __restrict__ xg, size_t bStride, int tStride,
               const __bf16* __restrict__ Whh,      // [4H, H] bf16
               __bf16* __restrict__ h0buf,          // [B,H] bf16 (parity 0)
               __bf16* __restrict__ h1buf,          // [B,H] bf16 (parity 1)
               float*  __restrict__ cstate,         // [B,H] f32
               __bf16* __restrict__ seq_out,        // [B,T,H] bf16
               int T, unsigned* bar_cnt, unsigned* bar_gen) {
  const int lane  = threadIdx.x & 31;
  const int wave  = threadIdx.x >> 5;        // 0..7
  const int gate  = wave >> 1;               // 0..3 (i,f,g,o)
  const int jhalf = wave & 1;                // which 16-wide j sub-tile
  const int b0 = blockIdx.x * 16;
  const int j0 = blockIdx.y * 32;
  const unsigned NWG = gridDim.x * gridDim.y;

  __shared__ float gates[4][16][32];

  // Preload this wave's Whh fragments for all K: rows [gate*H + j0 + jhalf*16, +16)
  const int wrow = gate * H_DIM + j0 + jhalf * 16;
  v16bf wfrag[8];
#pragma unroll
  for (int kk = 0; kk < 8; ++kk)
    wfrag[kk] = load_frag_16x32(Whh, wrow, H_DIM, kk * 32, lane);

  const int half = lane >> 4;
  const int cidx = lane & 15;
  const int ncol = wrow + cidx;              // gate-matrix column

  for (int t = 0; t < T; ++t) {
    const __bf16* hread  = (t & 1) ? h1buf : h0buf;
    __bf16*       hwrite = (t & 1) ? h0buf : h1buf;

    // prefetch next step's xg rows into L2 while we compute (CDNA5 prefetch)
    if (t + 1 < T && tStride != 0) {
#pragma unroll
      for (int r = 0; r < 8; ++r) {
        const int b = b0 + half * 8 + r;
        __builtin_prefetch(&xg[(size_t)b * bStride + (size_t)(t + 1) * tStride + ncol], 0, 1);
      }
    }

    // gates_tile = h[b0:b0+16, :] @ Whh[wrow : wrow+16, :]^T
    v8f acc = {};
#pragma unroll
    for (int kk = 0; kk < 8; ++kk) {
      v16bf a = load_frag_16x32(hread, b0, H_DIM, kk * 32, lane);
      acc = wmma_bf16(a, wfrag[kk], acc);
    }

    // add precomputed input projection and stash to LDS
#pragma unroll
    for (int r = 0; r < 8; ++r) {
      const int b = b0 + half * 8 + r;
      const float xv = (float)xg[(size_t)b * bStride + (size_t)t * tStride + ncol];
      gates[gate][half * 8 + r][jhalf * 16 + cidx] = acc[r] + xv;
    }
    __syncthreads();

    // LSTM cell update: 16x32 = 512 elements / 256 threads
    for (int e = threadIdx.x; e < 512; e += 256) {
      const int ml = e >> 5, jl = e & 31;
      const int b = b0 + ml, j = j0 + jl;
      const float iv = gates[0][ml][jl];
      const float fv = gates[1][ml][jl];
      const float gv = gates[2][ml][jl];
      const float ov = gates[3][ml][jl];
      const float cold = cstate[b * H_DIM + j];
      const float cnew = fast_sig(fv) * cold + fast_sig(iv) * fast_tanh(gv);
      const float hnew = fast_sig(ov) * fast_tanh(cnew);
      cstate[b * H_DIM + j] = cnew;
      const __bf16 hb = (__bf16)hnew;
      hwrite[b * H_DIM + j] = hb;
      seq_out[((size_t)b * T + t) * H_DIM + j] = hb;
    }

    // all WGs must see the new h before anyone starts step t+1
    grid_barrier(bar_cnt, bar_gen, NWG);
  }
}

// ---------------------------------------------------------------------------
// host side
// ---------------------------------------------------------------------------

extern "C" void kernel_launch(void* const* d_in, const int* in_sizes, int n_in,
                              void* d_out, int out_size, void* d_ws, size_t ws_size,
                              hipStream_t stream) {
  (void)in_sizes; (void)n_in; (void)out_size; (void)ws_size;
  const int B = B_DIM, T = T_DIM, D = D_DIM, H = H_DIM;

  const float* x = (const float*)d_in[0];
  // params: 1..4 enc0 (Wih,Whh,bih,bhh), 5..8 enc1, 9..12 dec0, 13..16 dec1
  const float* W_out = (const float*)d_in[17];
  const float* b_out = (const float*)d_in[18];
  float* out = (float*)d_out;

  // ---- carve workspace ----
  size_t off = 0;
  auto carve = [&](size_t bytes) -> void* {
    void* p = (char*)d_ws + off;
    off += (bytes + 255) & ~(size_t)255;
    return p;
  };
  __bf16* xb   = (__bf16*)carve((size_t)B * T * D * 2);       // x in bf16
  __bf16* seqA = (__bf16*)carve((size_t)B * T * H * 2);
  __bf16* seqB = (__bf16*)carve((size_t)B * T * H * 2);
  __bf16* xg   = (__bf16*)carve((size_t)B * T * G4H * 2);     // gate pre-proj

  const int in_dims[4] = {D, H, H, H};                        // enc0, enc1, dec0, dec1
  __bf16* w_ih[4]; __bf16* w_hh[4]; float* cbias[4];
  for (int l = 0; l < 4; ++l) {
    w_ih[l]  = (__bf16*)carve((size_t)G4H * in_dims[l] * 2);
    w_hh[l]  = (__bf16*)carve((size_t)G4H * H * 2);
    cbias[l] = (float*)carve((size_t)G4H * 4);
  }
  __bf16* woutb = (__bf16*)carve((size_t)D * H * 2);

  __bf16* hbuf[2][2]; float* cbuf[2];
  for (int s = 0; s < 2; ++s) {
    hbuf[s][0] = (__bf16*)carve((size_t)B * H * 2);
    hbuf[s][1] = (__bf16*)carve((size_t)B * H * 2);
    cbuf[s]    = (float*)carve((size_t)B * H * 4);
  }
  unsigned* bar = (unsigned*)carve(2 * sizeof(unsigned));

  // ---- init: barrier + zero states ----
  init_barrier<<<1, 32, 0, stream>>>(bar);
  for (int s = 0; s < 2; ++s) {
    zero_bf16<<<(B * H + 255) / 256, 256, 0, stream>>>(hbuf[s][0], B * H);
    zero_bf16<<<(B * H + 255) / 256, 256, 0, stream>>>(hbuf[s][1], B * H);
    zero_f32 <<<(B * H + 255) / 256, 256, 0, stream>>>(cbuf[s], B * H);
  }

  // ---- convert inputs / weights to bf16, combine biases ----
  {
    int n = B * T * D;
    cvt_f32_bf16<<<(n + 255) / 256, 256, 0, stream>>>(x, xb, n);
  }
  for (int l = 0; l < 4; ++l) {
    const float* Wih = (const float*)d_in[1 + 4 * l + 0];
    const float* Whh = (const float*)d_in[1 + 4 * l + 1];
    const float* bih = (const float*)d_in[1 + 4 * l + 2];
    const float* bhh = (const float*)d_in[1 + 4 * l + 3];
    int nih = G4H * in_dims[l], nhh = G4H * H;
    cvt_f32_bf16<<<(nih + 255) / 256, 256, 0, stream>>>(Wih, w_ih[l], nih);
    cvt_f32_bf16<<<(nhh + 255) / 256, 256, 0, stream>>>(Whh, w_hh[l], nhh);
    combine_bias<<<(G4H + 255) / 256, 256, 0, stream>>>(bih, bhh, cbias[l], G4H);
  }
  {
    int n = D * H;
    cvt_f32_bf16<<<(n + 255) / 256, 256, 0, stream>>>(W_out, woutb, n);
  }

  const dim3 gemmBlk(128);
  const dim3 scanBlk(256);
  const dim3 scanGrid(B / 16, H / 32);                        // 64 WGs
  const size_t bStrideSeq = (size_t)T * G4H;

  // ---- encoder layer 0 ----
  proj_gemm<false><<<dim3(B * T / 128, G4H / 32), gemmBlk, 0, stream>>>(
      xb, w_ih[0], cbias[0], xg, nullptr, D, G4H);
  lstm_scan<<<scanGrid, scanBlk, 0, stream>>>(xg, bStrideSeq, G4H, w_hh[0],
      hbuf[0][0], hbuf[0][1], cbuf[0], seqA, T, bar, bar + 1);

  // ---- encoder layer 1 ----
  proj_gemm<false><<<dim3(B * T / 128, G4H / 32), gemmBlk, 0, stream>>>(
      seqA, w_ih[1], cbias[1], xg, nullptr, H, G4H);
  lstm_scan<<<scanGrid, scanBlk, 0, stream>>>(xg, bStrideSeq, G4H, w_hh[1],
      hbuf[1][0], hbuf[1][1], cbuf[1], seqB, T, bar, bar + 1);

  // ---- decoder layer 0: broadcast input = enc final h (layer 1, parity 0) ----
  // single [B,4H] projection, read with timestep stride 0 in the scan
  proj_gemm<false><<<dim3(B / 128, G4H / 32), gemmBlk, 0, stream>>>(
      hbuf[1][0], w_ih[2], cbias[2], xg, nullptr, H, G4H);
  lstm_scan<<<scanGrid, scanBlk, 0, stream>>>(xg, (size_t)G4H, 0, w_hh[2],
      hbuf[0][0], hbuf[0][1], cbuf[0], seqA, T, bar, bar + 1);

  // ---- decoder layer 1 ----
  proj_gemm<false><<<dim3(B * T / 128, G4H / 32), gemmBlk, 0, stream>>>(
      seqA, w_ih[3], cbias[3], xg, nullptr, H, G4H);
  lstm_scan<<<scanGrid, scanBlk, 0, stream>>>(xg, bStrideSeq, G4H, w_hh[3],
      hbuf[1][0], hbuf[1][1], cbuf[1], seqB, T, bar, bar + 1);

  // ---- output projection: [B*T,H] x [D,H]^T + b_out -> f32 ----
  proj_gemm<true><<<dim3(B * T / 128, D / 32), gemmBlk, 0, stream>>>(
      seqB, woutb, b_out, nullptr, out, H, D);
}
